// GINE_84988812853437
// MI455X (gfx1250) — matrix operand order repaired and checked by reading, not compiled, not run on hardware
//
#include <hip/hip_runtime.h>

#define NN 20000
#define NE 320000
#define NG 256
#define RD 1536

typedef __attribute__((ext_vector_type(16))) __bf16 v16bf;
typedef __attribute__((ext_vector_type(8)))  float  v8f;

union Frag { uint4 u[2]; v16bf v; };

__device__ __forceinline__ unsigned short f2bf(float f) {
  unsigned u = __float_as_uint(f);
  u += 0x7FFFu + ((u >> 16) & 1u);          // round-to-nearest-even
  return (unsigned short)(u >> 16);
}

// async global->LDS 16B copy (CDNA5, tracked by ASYNCcnt)
__device__ __forceinline__ void async_cp16(unsigned ldsAddr, const void* gaddr) {
  asm volatile("global_load_async_to_lds_b128 %0, %1, off"
               :: "v"(ldsAddr), "v"(gaddr) : "memory");
}
__device__ __forceinline__ void wait_async0() {
  asm volatile("s_wait_asynccnt 0x0" ::: "memory");
}

// ---------------- generic f32 -> bf16 convert ----------------
__global__ void __launch_bounds__(256) convert_bf16_kernel(
    const float* __restrict__ src, unsigned short* __restrict__ dst, long n) {
  long i = (long)blockIdx.x * blockDim.x + threadIdx.x;
  long stride = (long)gridDim.x * blockDim.x;
  for (; i < n; i += stride) dst[i] = f2bf(src[i]);
}

// ------- f32 [K][N] -> bf16 transposed [N][K] (weights, done once per launch) -------
__global__ void __launch_bounds__(256) convertT_bf16_kernel(
    const float* __restrict__ src, unsigned short* __restrict__ dst, int K, int N) {
  long n = (long)K * N;
  long i = (long)blockIdx.x * blockDim.x + threadIdx.x;
  long stride = (long)gridDim.x * blockDim.x;
  for (; i < n; i += stride) {
    int k = (int)(i / N);
    int c = (int)(i - (long)k * N);
    dst[(size_t)c * K + k] = f2bf(src[i]);  // coalesced read, scattered write (tiny)
  }
}

// ---------------- t = (1+eps)*x + aggr  -> bf16 ----------------
__global__ void __launch_bounds__(256) combine_convert_kernel(
    const float* __restrict__ x, const float* __restrict__ aggr,
    const float* __restrict__ epsp, unsigned short* __restrict__ tb, long n) {
  float s = 1.0f + epsp[0];
  long i = (long)blockIdx.x * blockDim.x + threadIdx.x;
  long stride = (long)gridDim.x * blockDim.x;
  for (; i < n; i += stride) tb[i] = f2bf(fmaf(s, x[i], aggr[i]));
}

// ---- fused edge stage: e = ea@We+be ; msg = relu(x[src]+e) ; atomic scatter-add at dst ----
__global__ void __launch_bounds__(256) edge_fused_kernel(
    const float* __restrict__ x, const int* __restrict__ ei,
    const float* __restrict__ ea, const float* __restrict__ we,
    const float* __restrict__ be, float* __restrict__ aggr,
    int din, int nEdges) {
  extern __shared__ float smem[];
  float* weS = smem;            // [16*din]
  float* beS = smem + 16 * din; // [din]
  for (int i = threadIdx.x; i < 16 * din; i += blockDim.x) weS[i] = we[i];
  for (int i = threadIdx.x; i < din; i += blockDim.x) beS[i] = be[i];
  __syncthreads();
  const int lane = threadIdx.x & 31;
  const int gw = blockIdx.x * (blockDim.x >> 5) + (threadIdx.x >> 5);
  const int nw = gridDim.x * (blockDim.x >> 5);
  const int* __restrict__ srcI = ei;
  const int* __restrict__ dstI = ei + nEdges;
  const int dpl = din >> 5;                 // dims per lane (wave32)
  for (int e = gw; e < nEdges; e += nw) {
    int s = srcI[e], d = dstI[e];
    const float* __restrict__ eap = ea + (size_t)e * 16;
    float eav[16];
#pragma unroll
    for (int k = 0; k < 16; ++k) eav[k] = eap[k];   // same-address broadcast loads
    const float* __restrict__ xp = x + (size_t)s * din;
    float* __restrict__ ap = aggr + (size_t)d * din;
    for (int i = 0; i < dpl; ++i) {
      int dd = lane + (i << 5);
      float v = beS[dd];
#pragma unroll
      for (int k = 0; k < 16; ++k) v = fmaf(eav[k], weS[k * din + dd], v);
      v = fmaxf(v + xp[dd], 0.0f);
      atomicAdd(ap + dd, v);                // global_atomic_add_f32 @ L2
    }
  }
}

// -------- bf16 WMMA GEMM: block = 8 waves, tile 128(M) x 32(N); B pre-transposed [N][K] --------
// modes: 0 = GELU -> bf16 | 1 = BN+ReLU -> f32 + pooled atomicAdd | 2 = BN+ReLU -> bf16 | 3 = bias -> f32
__global__ void __launch_bounds__(256) wmma_gemm_kernel(
    const unsigned short* __restrict__ A, const unsigned short* __restrict__ BT,
    int M, int N, int K, const float* __restrict__ bias,
    const float* __restrict__ bng, const float* __restrict__ bnb,
    const float* __restrict__ bnm, const float* __restrict__ bnv,
    const int* __restrict__ batch, float* __restrict__ zpool, int zcol, int zld,
    void* __restrict__ outp, int mode) {
  __shared__ alignas(16) unsigned short aLds[128][40];  // +16B pad per row
  __shared__ alignas(16) unsigned short bLds[32][40];   // B rows = output columns, [n][k]
  const int tid  = threadIdx.x;
  const int wave = tid >> 5;
  const int lane = tid & 31;
  const int half = lane >> 4;
  const int l16  = lane & 15;
  const int bm0  = blockIdx.x * 128;
  const int bn0  = blockIdx.y * 32;
  v8f acc[2];
  acc[0] = (v8f){0.f, 0.f, 0.f, 0.f, 0.f, 0.f, 0.f, 0.f};
  acc[1] = (v8f){0.f, 0.f, 0.f, 0.f, 0.f, 0.f, 0.f, 0.f};

  for (int k0 = 0; k0 < K; k0 += 32) {
    // A tile 128x32 bf16: 512 x 16B chunks, async DMA; zero-fill tail rows via LDS store
    for (int c = tid; c < 512; c += 256) {
      int row = c >> 2;
      int k8  = (c & 3) << 3;
      int gm  = bm0 + row;
      unsigned la = (unsigned)(size_t)&aLds[row][k8];   // addr[31:0] = LDS offset
      if (gm < M) {
        async_cp16(la, A + (size_t)gm * K + k0 + k8);
      } else {
        uint4 zero = {0u, 0u, 0u, 0u};
        *(uint4*)(&aLds[row][k8]) = zero;
      }
    }
    // B tile 32x32 bf16 from pre-transposed weights: 128 x 16B chunks, async DMA
    for (int c = tid; c < 128; c += 256) {
      int row = c >> 2;
      int k8  = (c & 3) << 3;
      async_cp16((unsigned)(size_t)&bLds[row][k8],
                 BT + (size_t)(bn0 + row) * K + k0 + k8);
    }
    wait_async0();          // drain this wave's async DMAs into LDS
    __syncthreads();        // all waves' tiles visible

    // A frag (contiguous k order, same as B): half 0 -> K 0..15 ; half 1 -> K 16..31
    Frag fa, fb0, fb1;
    const unsigned short* ar = &aLds[wave * 16 + l16][0];
    fa.u[0] = *(const uint4*)(ar + 16 * half);
    fa.u[1] = *(const uint4*)(ar + 16 * half + 8);
    // B frag: lanes 0-15 K=0..15, lanes 16-31 K=16..31 (contiguous per row in [n][k])
    const unsigned short* br0 = &bLds[l16][0];
    fb0.u[0] = *(const uint4*)(br0 + 16 * half);
    fb0.u[1] = *(const uint4*)(br0 + 16 * half + 8);
    const unsigned short* br1 = &bLds[16 + l16][0];
    fb1.u[0] = *(const uint4*)(br1 + 16 * half);
    fb1.u[1] = *(const uint4*)(br1 + 16 * half + 8);

    acc[0] = __builtin_amdgcn_wmma_f32_16x16x32_bf16(false, fa.v, false, fb0.v,
                                                     (short)0, acc[0], false, false);
    acc[1] = __builtin_amdgcn_wmma_f32_16x16x32_bf16(false, fa.v, false, fb1.v,
                                                     (short)0, acc[1], false, false);
    __syncthreads();
  }

  // D layout: element r -> tile row (r + 8*half), col (lane&15)
#pragma unroll
  for (int t = 0; t < 2; ++t) {
    int n_g = bn0 + t * 16 + l16;
    float bv = bias ? bias[n_g] : 0.0f;
    float scale = 1.0f, shift = 0.0f;
    if (mode == 1 || mode == 2) {
      float g = bng[n_g], bb = bnb[n_g], mm = bnm[n_g], vv = bnv[n_g];
      scale = g * rsqrtf(vv + 1e-5f);
      shift = bb - mm * scale;
    }
#pragma unroll
    for (int r = 0; r < 8; ++r) {
      int row = bm0 + wave * 16 + half * 8 + r;
      if (row >= M) continue;
      float v = acc[t][r] + bv;
      if (mode == 0) {
        v = 0.5f * v * (1.0f + erff(v * 0.70710678118654752f));   // exact GELU
        ((unsigned short*)outp)[(size_t)row * N + n_g] = f2bf(v);
      } else if (mode == 1) {
        v = fmaxf(fmaf(v, scale, shift), 0.0f);
        ((float*)outp)[(size_t)row * N + n_g] = v;
        atomicAdd(&zpool[(size_t)batch[row] * zld + zcol + n_g], v); // fused pooling
      } else if (mode == 2) {
        v = fmaxf(fmaf(v, scale, shift), 0.0f);
        ((unsigned short*)outp)[(size_t)row * N + n_g] = f2bf(v);
      } else {
        ((float*)outp)[(size_t)row * N + n_g] = v;
      }
    }
  }
}

extern "C" void kernel_launch(void* const* d_in, const int* in_sizes, int n_in,
                              void* d_out, int out_size, void* d_ws, size_t ws_size,
                              hipStream_t stream) {
  (void)in_sizes; (void)n_in; (void)out_size; (void)ws_size;
  const float* x0  = (const float*)d_in[0];
  const int*   ei  = (const int*)d_in[1];    // JAX x64 disabled -> int32
  const float* ea  = (const float*)d_in[2];
  const int*   bat = (const int*)d_in[3];

  static const int IN[5]   = {128, 256, 256, 256, 256};
  static const int OUT[5]  = {256, 256, 256, 256, 512};
  static const int ZOFF[5] = {0, 256, 512, 768, 1024};

  char* ws = (char*)d_ws;
  size_t off = 0;
  auto alloc = [&](size_t b) -> void* {
    void* p = ws + off;
    off = (off + b + 255) & ~(size_t)255;
    return p;
  };
  float*          nodeH = (float*)alloc((size_t)NN * 512 * 4);
  float*          aggr  = (float*)alloc((size_t)NN * 256 * 4);
  unsigned short* tb    = (unsigned short*)alloc((size_t)NN * 256 * 2);
  unsigned short* hmidb = (unsigned short*)alloc((size_t)NN * 512 * 2);
  unsigned short* wb1   = (unsigned short*)alloc((size_t)1536 * 1024 * 2);
  unsigned short* wb2   = (unsigned short*)alloc((size_t)512 * 512 * 2);
  float*          z     = (float*)alloc((size_t)NG * RD * 4);
  unsigned short* zb    = (unsigned short*)alloc((size_t)NG * RD * 2);
  unsigned short* zhb   = (unsigned short*)alloc((size_t)NG * 1024 * 2);

  hipMemsetAsync(z, 0, (size_t)NG * RD * 4, stream);

  const float* curx = x0;
  for (int l = 0; l < 5; ++l) {
    int din = IN[l], dout = OUT[l];
    int base = 4 + 11 * l;
    const float* we  = (const float*)d_in[base + 0];
    const float* be  = (const float*)d_in[base + 1];
    const float* w1  = (const float*)d_in[base + 2];
    const float* b1  = (const float*)d_in[base + 3];
    const float* w2  = (const float*)d_in[base + 4];
    const float* b2  = (const float*)d_in[base + 5];
    const float* eps = (const float*)d_in[base + 6];
    const float* bng = (const float*)d_in[base + 7];
    const float* bnb = (const float*)d_in[base + 8];
    const float* bnm = (const float*)d_in[base + 9];
    const float* bnv = (const float*)d_in[base + 10];

    hipMemsetAsync(aggr, 0, (size_t)NN * din * 4, stream);
    edge_fused_kernel<<<2048, 256, (size_t)(17 * din) * 4, stream>>>(
        curx, ei, ea, we, be, aggr, din, NE);
    combine_convert_kernel<<<4096, 256, 0, stream>>>(curx, aggr, eps, tb, (long)NN * din);
    convertT_bf16_kernel<<<512, 256, 0, stream>>>(w1, wb1, din, dout);   // -> [dout][din]
    convertT_bf16_kernel<<<512, 256, 0, stream>>>(w2, wb2, dout, dout);  // -> [dout][dout]

    dim3 g1((NN + 127) / 128, dout / 32);
    // hmid = gelu(t @ W1 + b1)  -> bf16
    wmma_gemm_kernel<<<g1, 256, 0, stream>>>(tb, wb1, NN, dout, din, b1,
        nullptr, nullptr, nullptr, nullptr, nullptr, nullptr, 0, 0, hmidb, 0);
    // h = relu(bn(hmid @ W2 + b2)) -> f32 nodeH, fused pooled segment_sum into z
    wmma_gemm_kernel<<<g1, 256, 0, stream>>>(hmidb, wb2, NN, dout, dout, b2,
        bng, bnb, bnm, bnv, bat, z, ZOFF[l], RD, nodeH, 1);
    curx = nodeH;   // x fully consumed before the GEMMs run; safe in-place
  }

  // classifier: z[256x1536] @ W1 -> BN -> ReLU -> @ W2 -> logits[256x128]
  const float* cw1  = (const float*)d_in[59];
  const float* cb1  = (const float*)d_in[60];
  const float* cbng = (const float*)d_in[61];
  const float* cbnb = (const float*)d_in[62];
  const float* cbnm = (const float*)d_in[63];
  const float* cbnv = (const float*)d_in[64];
  const float* cw2  = (const float*)d_in[65];
  const float* cb2  = (const float*)d_in[66];

  convert_bf16_kernel<<<512, 256, 0, stream>>>(z, zb, (long)NG * RD);
  convertT_bf16_kernel<<<2048, 256, 0, stream>>>(cw1, wb1, RD, 1024);   // -> [1024][1536]
  convertT_bf16_kernel<<<512, 256, 0, stream>>>(cw2, wb2, 1024, 128);   // -> [128][1024]

  dim3 g2((NG + 127) / 128, 1024 / 32);
  wmma_gemm_kernel<<<g2, 256, 0, stream>>>(zb, wb1, NG, 1024, RD, cb1,
      cbng, cbnb, cbnm, cbnv, nullptr, nullptr, 0, 0, zhb, 2);
  dim3 g3((NG + 127) / 128, 128 / 32);
  wmma_gemm_kernel<<<g3, 256, 0, stream>>>(zhb, wb2, NG, 128, 1024, cb2,
      nullptr, nullptr, nullptr, nullptr, nullptr, nullptr, 0, 0, (float*)d_out, 3);
}
